// Receiver_26998164422995
// MI455X (gfx1250) — matrix operand order repaired and checked by reading, not compiled, not run on hardware
//
#include <hip/hip_runtime.h>

// ---------------- problem constants (from reference) ----------------
#define N_NODES 100000
#define N_EDGES 1600000
#define N_ETOT  (N_EDGES + N_NODES)   // with self loops
#define F_IN    128
#define EMB     128
#define HEADS   4
#define D_HEAD  32
#define HIDDEN  256
#define BATCH   2048
#define CANDS   16

typedef __attribute__((ext_vector_type(16))) __bf16 v16bf;
typedef __attribute__((ext_vector_type(8)))  float  v8f;
typedef __attribute__((ext_vector_type(4)))  int    v4i;

struct Frag {
    union {
        unsigned short u[16];
        uint4          q[2];
        v16bf          v;
    };
};

// detect gfx1250 async global->LDS builtins (guarded; fallback = plain copy)
#if defined(__has_builtin)
#  if __has_builtin(__builtin_amdgcn_global_load_async_to_lds_b128)
#    define HAVE_ASYNC_LDS 1
#  endif
#  if __has_builtin(__builtin_amdgcn_s_wait_asynccnt)
#    define HAVE_WAIT_ASYNC 1
#  endif
#endif

// float -> bf16 round-to-nearest-even
__device__ __forceinline__ unsigned short f2bf(float f) {
    unsigned int u = __float_as_uint(f);
    unsigned int r = u + 0x7FFFu + ((u >> 16) & 1u);
    return (unsigned short)(r >> 16);
}

// monotone float<->uint key (for atomic max on floats of any sign)
__device__ __forceinline__ unsigned int fkey(float v) {
    unsigned int u = __float_as_uint(v);
    return (u & 0x80000000u) ? ~u : (u | 0x80000000u);
}
__device__ __forceinline__ float funkey(unsigned int k) {
    unsigned int b = (k & 0x80000000u) ? (k & 0x7FFFFFFFu) : ~k;
    return __uint_as_float(b);
}
#define KEY_NEG_INF 0x007FFFFFu   // fkey(-inf)

// ---------------- tiny utility kernels ----------------
__global__ void k_f2bf(const float* __restrict__ s, unsigned short* __restrict__ d, int n) {
    int t = blockIdx.x * blockDim.x + threadIdx.x;
    if (t < n) d[t] = f2bf(s[t]);
}

// wT[n*F_IN + k] = bf16(W[k*EMB + n])  : transposed (N-major) B operand for GEMM1
__global__ void k_w_transpose(const float* __restrict__ W, unsigned short* __restrict__ wT) {
    int t = blockIdx.x * blockDim.x + threadIdx.x;
    if (t >= F_IN * EMB) return;
    int k = t & (F_IN - 1);
    int n = t >> 7;
    wT[t] = f2bf(W[k * EMB + n]);
}

__global__ void k_memset32(unsigned int* __restrict__ p, unsigned int v, int n) {
    int t = blockIdx.x * blockDim.x + threadIdx.x;
    if (t < n) p[t] = v;
}

// ---------------- GEMM 1: Wx = x @ W  (bf16 WMMA, f32 accum) ----------------
// block = 256 threads = 8 waves; block -> one M-tile row, wave -> one N-tile.
// B operand staged in LDS as W^T (N-major) so every fragment is 2 contiguous b128 loads.
__global__ __launch_bounds__(256) void k_gemm_wx(const unsigned short* __restrict__ xbf,
                                                 const unsigned short* __restrict__ wT,
                                                 float* __restrict__ Wx) {
    __shared__ __align__(16) unsigned short ldsWT[F_IN * EMB];   // 32 KB of 320 KB WGP LDS
#ifdef HAVE_ASYNC_LDS
    {   // gfx1250 async DMA: global -> LDS, 16B per op, tracked by ASYNCcnt
        typedef __attribute__((address_space(1))) v4i* g4p;   // global int4*
        typedef __attribute__((address_space(3))) v4i* l4p;   // LDS int4*
        unsigned short* wt_nc = const_cast<unsigned short*>(wT);
        for (int i = threadIdx.x; i < (F_IN * EMB) / 8; i += 256) {
            g4p g = (g4p)(wt_nc + (size_t)i * 8);
            l4p l = (l4p)(ldsWT + (size_t)i * 8);
            __builtin_amdgcn_global_load_async_to_lds_b128(g, l, 0, 0);
        }
#  ifdef HAVE_WAIT_ASYNC
        __builtin_amdgcn_s_wait_asynccnt(0);
#  else
        asm volatile("s_wait_asynccnt 0" ::: "memory");
#  endif
    }
#else
    {   // fallback: vector copy through VGPRs
        const uint4* s = (const uint4*)wT;
        uint4* d = (uint4*)ldsWT;
        for (int i = threadIdx.x; i < (F_IN * EMB) / 8; i += 256) d[i] = s[i];
    }
#endif
    __syncthreads();

    const int lane = threadIdx.x & 31;
    const int wave = threadIdx.x >> 5;
    const int mt = blockIdx.x;          // 6250 tiles of 16 rows
    const int nt = wave;                // 8 tiles of 16 cols
    const int mrow = mt * 16 + (lane & 15);
    const int ncol = nt * 16 + (lane & 15);
    const int kbA = (lane < 16) ? 0 : 8;
    const int kbB = (lane < 16) ? 0 : 8;

    v8f c = {};
#pragma unroll
    for (int k0 = 0; k0 < F_IN; k0 += 32) {
        Frag a, b;
        const uint4* ap = (const uint4*)(xbf + (size_t)mrow * F_IN + k0 + kbA);
        __builtin_prefetch(ap + 4, 0, 3);                 // global_prefetch_b8, next K chunk
        a.q[0] = ap[0];                                   // K = k0+kbA+0..7
        a.q[1] = ap[2];                                   // K = k0+kbA+16..23
        const uint4* bp = (const uint4*)(ldsWT + (size_t)ncol * F_IN + k0 + kbB);
        b.q[0] = bp[0];                                   // K = k0+kbB+0..7   (ds_load_b128)
        b.q[1] = bp[2];                                   // K = k0+kbB+16..23
        c = __builtin_amdgcn_wmma_f32_16x16x32_bf16(false, a.v, false, b.v,
                                                    (short)0, c, false, false);
    }
    const int rbase = mt * 16 + ((lane < 16) ? 0 : 8);
#pragma unroll
    for (int r = 0; r < 8; ++r)
        Wx[(size_t)(rbase + r) * EMB + ncol] = c[r];
}

// ---------------- attention coefficients a_s, a_d : [N, H] ----------------
__global__ void k_attn(const float* __restrict__ Wx, const float* __restrict__ att_src,
                       const float* __restrict__ att_dst,
                       float* __restrict__ a_s, float* __restrict__ a_d) {
    int t = blockIdx.x * blockDim.x + threadIdx.x;
    if (t >= N_NODES * HEADS) return;
    int n = t >> 2, h = t & 3;
    const float* w = Wx + (size_t)n * EMB + h * D_HEAD;
    const float* s = att_src + h * D_HEAD;
    const float* d = att_dst + h * D_HEAD;
    float vs = 0.f, vd = 0.f;
#pragma unroll 8
    for (int j = 0; j < D_HEAD; ++j) { float x = w[j]; vs += x * s[j]; vd += x * d[j]; }
    a_s[t] = vs; a_d[t] = vd;
}

// ---------------- edge pass helpers ----------------
__device__ __forceinline__ void edge_endpoints(int e, const int* __restrict__ ei, int& s, int& d) {
    if (e < N_EDGES) { s = ei[e]; d = ei[N_EDGES + e]; }
    else             { s = d = e - N_EDGES; }            // appended self loops
}

// pass 1: segment max (uint-key atomic max -> global_atomic_max_u32)
__global__ void k_edge_max(const int* __restrict__ ei, const float* __restrict__ a_s,
                           const float* __restrict__ a_d, unsigned int* __restrict__ mkey) {
    int t = blockIdx.x * blockDim.x + threadIdx.x;
    if (t >= N_ETOT * HEADS) return;
    int e = t >> 2, h = t & 3, s, d;
    edge_endpoints(e, ei, s, d);
    float v = a_s[s * 4 + h] + a_d[d * 4 + h];
    v = v > 0.f ? v : 0.2f * v;                          // leaky_relu(0.2)
    atomicMax(&mkey[d * 4 + h], fkey(v));
}

// pass 2: softmax denominator (global_atomic_add_f32, L2-resident)
__global__ void k_edge_den(const int* __restrict__ ei, const float* __restrict__ a_s,
                           const float* __restrict__ a_d, const unsigned int* __restrict__ mkey,
                           float* __restrict__ den) {
    int t = blockIdx.x * blockDim.x + threadIdx.x;
    if (t >= N_ETOT * HEADS) return;
    int e = t >> 2, h = t & 3, s, d;
    edge_endpoints(e, ei, s, d);
    float v = a_s[s * 4 + h] + a_d[d * 4 + h];
    v = v > 0.f ? v : 0.2f * v;
    float ex = expf(v - funkey(mkey[d * 4 + h]));
    __hip_atomic_fetch_add(&den[d * 4 + h], ex, __ATOMIC_RELAXED, __HIP_MEMORY_SCOPE_AGENT);
}

// pass 3: weighted aggregation, one wave per edge (4 floats / lane)
__global__ __launch_bounds__(256) void k_edge_agg(const int* __restrict__ ei,
                                                  const float* __restrict__ a_s,
                                                  const float* __restrict__ a_d,
                                                  const unsigned int* __restrict__ mkey,
                                                  const float* __restrict__ den,
                                                  const float* __restrict__ Wx,
                                                  float* __restrict__ outg) {
    int wid = (blockIdx.x * 256 + threadIdx.x) >> 5;
    int lane = threadIdx.x & 31;
    if (wid >= N_ETOT) return;
    int s, d;
    edge_endpoints(wid, ei, s, d);
    int h = lane >> 3;                                   // 4 consecutive elems stay in one head
    float v = a_s[s * 4 + h] + a_d[d * 4 + h];
    v = v > 0.f ? v : 0.2f * v;
    float alpha = expf(v - funkey(mkey[d * 4 + h])) / den[d * 4 + h];
    const float4 w4 = *(const float4*)(Wx + (size_t)s * EMB + lane * 4);
    float* od = outg + (size_t)d * EMB + lane * 4;
    __hip_atomic_fetch_add(od + 0, alpha * w4.x, __ATOMIC_RELAXED, __HIP_MEMORY_SCOPE_AGENT);
    __hip_atomic_fetch_add(od + 1, alpha * w4.y, __ATOMIC_RELAXED, __HIP_MEMORY_SCOPE_AGENT);
    __hip_atomic_fetch_add(od + 2, alpha * w4.z, __ATOMIC_RELAXED, __HIP_MEMORY_SCOPE_AGENT);
    __hip_atomic_fetch_add(od + 3, alpha * w4.w, __ATOMIC_RELAXED, __HIP_MEMORY_SCOPE_AGENT);
}

// ---------------- GEMM 2: msg = message @ fc_w^T + fc_b (bf16 WMMA) ----------------
// B = fc_w^T  =>  B^T = fc_w, so bf16(fc_w) IS the fragment-ready (N-major) layout.
__global__ __launch_bounds__(256) void k_gemm_msg(const unsigned short* __restrict__ mbf,
                                                  const unsigned short* __restrict__ fcwbf,
                                                  const float* __restrict__ fc_b,
                                                  float* __restrict__ msg) {
    const int lane = threadIdx.x & 31;
    const int wave = threadIdx.x >> 5;
    const int mt = blockIdx.x;          // 128 M-tiles
    const int nt = wave;                // 8 N-tiles
    const int mrow = mt * 16 + (lane & 15);
    const int ncol = nt * 16 + (lane & 15);
    const int kbA = (lane < 16) ? 0 : 8;
    const int kbB = (lane < 16) ? 0 : 8;

    v8f c = {};
#pragma unroll
    for (int k0 = 0; k0 < HIDDEN; k0 += 32) {
        Frag a, b;
        const uint4* ap = (const uint4*)(mbf + (size_t)mrow * HIDDEN + k0 + kbA);
        __builtin_prefetch(ap + 4, 0, 3);
        a.q[0] = ap[0];
        a.q[1] = ap[2];
        const uint4* bp = (const uint4*)(fcwbf + (size_t)ncol * HIDDEN + k0 + kbB);
        b.q[0] = bp[0];
        b.q[1] = bp[2];
        c = __builtin_amdgcn_wmma_f32_16x16x32_bf16(false, a.v, false, b.v,
                                                    (short)0, c, false, false);
    }
    const int rbase = mt * 16 + ((lane < 16) ? 0 : 8);
#pragma unroll
    for (int r = 0; r < 8; ++r)
        msg[(size_t)(rbase + r) * EMB + ncol] = c[r] + fc_b[ncol];
}

// ---------------- candidate scoring: one wave per (b, c) ----------------
__global__ __launch_bounds__(256) void k_score(const int* __restrict__ idx,
                                               const float* __restrict__ outg,
                                               const float* __restrict__ bias,
                                               const float* __restrict__ msg,
                                               float* __restrict__ dots,
                                               unsigned int* __restrict__ amax) {
    int wid = (blockIdx.x * 256 + threadIdx.x) >> 5;
    int lane = threadIdx.x & 31;
    if (wid >= BATCH * CANDS) return;
    int b = wid >> 4;
    int node = idx[wid];
    const float* e = outg + (size_t)node * EMB + lane * 4;
    const float* m = msg + (size_t)b * EMB + lane * 4;
    const float* bi = bias + lane * 4;
    float s = 0.f;
#pragma unroll
    for (int j = 0; j < 4; ++j) s += (e[j] + bi[j]) * m[j];
#pragma unroll
    for (int off = 16; off; off >>= 1) s += __shfl_xor(s, off, 32);
    if (lane == 0) {
        dots[wid] = s;
        atomicMax(amax, __float_as_uint(fabsf(s)));      // |x| >= 0 -> plain uint order
    }
}

// ---------------- per-row log_softmax over 16 candidates ----------------
__global__ void k_logsoftmax(const float* __restrict__ dots, float* __restrict__ out) {
    int b = blockIdx.x * blockDim.x + threadIdx.x;
    if (b >= BATCH) return;
    float d[CANDS];
    float mx = -1e30f;
#pragma unroll
    for (int c = 0; c < CANDS; ++c) { d[c] = dots[b * CANDS + c]; mx = fmaxf(mx, d[c]); }
    float sum = 0.f;
#pragma unroll
    for (int c = 0; c < CANDS; ++c) sum += expf(d[c] - mx);
    float l = logf(sum);
#pragma unroll
    for (int c = 0; c < CANDS; ++c) out[b * CANDS + c] = d[c] - mx - l;
}

// ---------------- launch ----------------
static inline unsigned cdiv(size_t a, unsigned b) { return (unsigned)((a + b - 1) / b); }

extern "C" void kernel_launch(void* const* d_in, const int* in_sizes, int n_in,
                              void* d_out, int out_size, void* d_ws, size_t ws_size,
                              hipStream_t stream) {
    const float* message = (const float*)d_in[0];   // [2048, 256]
    const float* x       = (const float*)d_in[1];   // [100000, 128]
    const int*   ei      = (const int*)d_in[2];     // [2, 1600000]
    const int*   indices = (const int*)d_in[3];     // [32768]
    const float* W       = (const float*)d_in[4];   // [128, 4, 32] -> [128, 128]
    const float* att_src = (const float*)d_in[5];   // [4, 32]
    const float* att_dst = (const float*)d_in[6];   // [4, 32]
    const float* bias    = (const float*)d_in[7];   // [128]
    const float* fc_w    = (const float*)d_in[8];   // [128, 256]
    const float* fc_b    = (const float*)d_in[9];   // [128]
    float* out = (float*)d_out;                     // [2048, 16]

    // workspace carve-out (256B aligned)
    char* wsb = (char*)d_ws;
    size_t off = 0;
    auto carve = [&](size_t bytes) -> char* {
        char* p = wsb + off;
        off = (off + bytes + 255) & ~(size_t)255;
        return p;
    };
    unsigned short* xbf   = (unsigned short*)carve((size_t)N_NODES * F_IN * 2);
    unsigned short* wT    = (unsigned short*)carve((size_t)F_IN * EMB * 2);
    unsigned short* fcwbf = (unsigned short*)carve((size_t)EMB * HIDDEN * 2);
    unsigned short* mbf   = (unsigned short*)carve((size_t)BATCH * HIDDEN * 2);
    float* Wx   = (float*)carve((size_t)N_NODES * EMB * 4);
    float* a_s  = (float*)carve((size_t)N_NODES * HEADS * 4);
    float* a_d  = (float*)carve((size_t)N_NODES * HEADS * 4);
    unsigned int* mkey = (unsigned int*)carve((size_t)N_NODES * HEADS * 4);
    float* den  = (float*)carve((size_t)N_NODES * HEADS * 4);
    float* outg = (float*)carve((size_t)N_NODES * EMB * 4);
    float* msg  = (float*)carve((size_t)BATCH * EMB * 4);
    float* dots = (float*)carve((size_t)BATCH * CANDS * 4);
    unsigned int* amax = (unsigned int*)carve(256);

    // precision conversions / operand layouts
    k_f2bf<<<cdiv((size_t)N_NODES * F_IN, 256), 256, 0, stream>>>(x, xbf, N_NODES * F_IN);
    k_w_transpose<<<cdiv((size_t)F_IN * EMB, 256), 256, 0, stream>>>(W, wT);
    k_f2bf<<<cdiv((size_t)EMB * HIDDEN, 256), 256, 0, stream>>>(fc_w, fcwbf, EMB * HIDDEN);
    k_f2bf<<<cdiv((size_t)BATCH * HIDDEN, 256), 256, 0, stream>>>(message, mbf, BATCH * HIDDEN);

    // accumulator init
    k_memset32<<<cdiv((size_t)N_NODES * EMB, 256), 256, 0, stream>>>((unsigned int*)outg, 0u, N_NODES * EMB);
    k_memset32<<<cdiv((size_t)N_NODES * HEADS, 256), 256, 0, stream>>>((unsigned int*)den, 0u, N_NODES * HEADS);
    k_memset32<<<cdiv((size_t)N_NODES * HEADS, 256), 256, 0, stream>>>(mkey, KEY_NEG_INF, N_NODES * HEADS);
    k_memset32<<<1, 32, 0, stream>>>(amax, 0u, 1);

    // GAT
    k_gemm_wx<<<N_NODES / 16, 256, 0, stream>>>(xbf, wT, Wx);
    k_attn<<<cdiv((size_t)N_NODES * HEADS, 256), 256, 0, stream>>>(Wx, att_src, att_dst, a_s, a_d);
    k_edge_max<<<cdiv((size_t)N_ETOT * HEADS, 256), 256, 0, stream>>>(ei, a_s, a_d, mkey);
    k_edge_den<<<cdiv((size_t)N_ETOT * HEADS, 256), 256, 0, stream>>>(ei, a_s, a_d, mkey, den);
    k_edge_agg<<<cdiv((size_t)N_ETOT * 32, 256), 256, 0, stream>>>(ei, a_s, a_d, mkey, den, Wx, outg);

    // scoring head
    k_gemm_msg<<<BATCH / 16, 256, 0, stream>>>(mbf, fcwbf, fc_b, msg);
    k_score<<<cdiv((size_t)BATCH * CANDS * 32, 256), 256, 0, stream>>>(indices, outg, bias, msg, dots, amax);
    k_logsoftmax<<<cdiv(BATCH, 256), 256, 0, stream>>>(dots, out);

    (void)in_sizes; (void)n_in; (void)out_size; (void)ws_size;
}
